// AdaLNBlock_89970974916610
// MI455X (gfx1250) — matrix operand order repaired and checked by reading, not compiled, not run on hardware
//
#include <hip/hip_runtime.h>
#include <hip/hip_bf16.h>

// ---------------------------------------------------------------------------
// AdaLN transformer block for MI455X (gfx1250, wave32, WMMA).
// GEMMs + attention use V_WMMA_F32_16X16X32_BF16 (bf16 in, f32 acc).
// Tile staging uses CDNA5 async global->LDS DMA (ASYNCcnt) + prefetch.
// ---------------------------------------------------------------------------

typedef __attribute__((ext_vector_type(16))) __bf16 bf16x16;
typedef __attribute__((ext_vector_type(8)))  float  floatx8;

union FragU {
    bf16x16 v;
    uint4   q[2];
    unsigned short u[16];
};

__device__ __forceinline__ unsigned short f32_to_bf16(float f) {
    unsigned int u = __builtin_bit_cast(unsigned int, f);
    u += 0x7FFFu + ((u >> 16) & 1u);   // round-to-nearest-even
    return (unsigned short)(u >> 16);
}

// flat LDS address low 32 bits == byte offset from LDS_BASE (ISA aperture rule)
__device__ __forceinline__ unsigned lds_off(const void* p) {
    return (unsigned)(unsigned long long)p;
}

// CDNA5 async DMA: global -> LDS, 16B per lane, tracked by ASYNCcnt.
__device__ __forceinline__ void async_load_b128(const void* gptr, unsigned loff) {
    asm volatile("global_load_async_to_lds_b128 %0, %1, off"
                 :: "v"(loff), "v"(gptr) : "memory");
}
#define WAIT_ASYNCCNT(n) asm volatile("s_wait_asynccnt " #n ::: "memory")

// ---------------------------------------------------------------------------
// Elementwise helpers
// ---------------------------------------------------------------------------
__global__ void f2bf_kernel(const float* __restrict__ in,
                            unsigned short* __restrict__ out, int n) {
    int i = blockIdx.x * 256 + threadIdx.x;
    if (i < n) out[i] = f32_to_bf16(in[i]);
}

__global__ void silu_bf16_kernel(const float* __restrict__ in,
                                 unsigned short* __restrict__ out, int n) {
    int i = blockIdx.x * 256 + threadIdx.x;
    if (i < n) {
        float v = in[i];
        out[i] = f32_to_bf16(v / (1.0f + __expf(-v)));
    }
}

// AdaLN: out_bf16 = LN(x) * (1 + mod[scale_off..]) + mod[shift_off..]
// one wave per token (512 elems -> 16 per lane), 8 waves / block.
__global__ __launch_bounds__(256) void adaln_kernel(
    const float* __restrict__ xin, const float* __restrict__ mod,
    int scale_off, int shift_off, unsigned short* __restrict__ outbf)
{
    const int lane = threadIdx.x & 31;
    const int wave = threadIdx.x >> 5;
    const size_t tok = (size_t)blockIdx.x * 8 + wave;
    const float* xr = xin + tok * 512 + lane * 16;

    float x[16];
    *(float4*)&x[0]  = *(const float4*)(xr);
    *(float4*)&x[4]  = *(const float4*)(xr + 4);
    *(float4*)&x[8]  = *(const float4*)(xr + 8);
    *(float4*)&x[12] = *(const float4*)(xr + 12);

    float s = 0.f, s2 = 0.f;
    #pragma unroll
    for (int e = 0; e < 16; ++e) { s += x[e]; s2 += x[e] * x[e]; }
    #pragma unroll
    for (int off = 16; off >= 1; off >>= 1) {
        s  += __shfl_xor(s,  off);
        s2 += __shfl_xor(s2, off);
    }
    float mu  = s * (1.0f / 512.0f);
    float var = s2 * (1.0f / 512.0f) - mu * mu;
    float r   = rsqrtf(var + 1e-5f);

    const float* sc = mod + tok * 3072 + scale_off + lane * 16;
    const float* sh = mod + tok * 3072 + shift_off + lane * 16;
    __align__(16) unsigned short o[16];
    #pragma unroll
    for (int e = 0; e < 16; ++e)
        o[e] = f32_to_bf16((x[e] - mu) * r * (1.0f + sc[e]) + sh[e]);
    unsigned short* op = outbf + tok * 512 + lane * 16;
    *(uint4*)(op)     = *(const uint4*)&o[0];
    *(uint4*)(op + 8) = *(const uint4*)&o[8];
}

// ---------------------------------------------------------------------------
// Generic WMMA GEMM: C[M,N] = A[M,K](bf16) * W[N,K](bf16)^T  (+ epilogue)
// block tile 128x128, 8 waves (4M x 2N), wave tile 32x64 (2x4 WMMA tiles),
// K-step 32. Double-buffered LDS, async global->LDS staging, peeled steady
// state (no branches in the hot loop), branchless clamped prefetch of k+2.
// mode 0: mod-proj   -> f32  (acc + bias)
// mode 1: qkv        -> bf16 (acc)
// mode 2: out-proj   -> f32  resid + gate*(acc+bias)
// mode 3: mlp-up     -> bf16 gelu(acc+bias)
// mode 4: mlp-down   -> f32  resid + gate*(acc+bias)
// ---------------------------------------------------------------------------
__global__ __launch_bounds__(256) void gemm_bf16_kernel(
    const unsigned short* __restrict__ A, const unsigned short* __restrict__ W,
    const float* __restrict__ bias, int K, int N, int mode,
    const float* __restrict__ resid, const float* __restrict__ mod, int modoff,
    float* __restrict__ outf, unsigned short* __restrict__ outbf)
{
    __shared__ unsigned short Alds[2][128 * 40];   // +8 halves pad / row
    __shared__ unsigned short Wlds[2][128 * 40];

    const int tid   = threadIdx.x;
    const int lane  = tid & 31;
    const int wave  = tid >> 5;
    const int waveM = wave & 3;    // 4 M-groups of 32 rows
    const int waveN = wave >> 2;   // 2 N-groups of 64 cols
    const int bm = blockIdx.x * 128;
    const int bn = blockIdx.y * 128;

    floatx8 acc[2][4];
    #pragma unroll
    for (int i = 0; i < 2; ++i)
        #pragma unroll
        for (int j = 0; j < 4; ++j) acc[i][j] = {};

    // issue one tile's async loads: 2x A + 2x W per thread (4 in flight)
    auto stage = [&](int buf, int kk) {
        const int k0 = kk << 5;
        #pragma unroll
        for (int rp = 0; rp < 2; ++rp) {
            int i = rp * 256 + tid;
            int row = i >> 2, seg = i & 3;
            async_load_b128(&A[(size_t)(bm + row) * K + k0 + seg * 8],
                            lds_off(&Alds[buf][row * 40 + seg * 8]));
            async_load_b128(&W[(size_t)(bn + row) * K + k0 + seg * 8],
                            lds_off(&Wlds[buf][row * 40 + seg * 8]));
        }
    };

    // one k-step of WMMA work out of buffer `cur`
    auto compute = [&](int cur) {
        FragU af[2], wf[4];
        #pragma unroll
        for (int sm = 0; sm < 2; ++sm) {
            int arow = waveM * 32 + sm * 16 + (lane & 15);
            int base = (lane >> 4) * 8;                 // A layout: two 8-half runs
            af[sm].q[0] = *(const uint4*)&Alds[cur][arow * 40 + base];
            af[sm].q[1] = *(const uint4*)&Alds[cur][arow * 40 + base + 16];
        }
        #pragma unroll
        for (int sn = 0; sn < 4; ++sn) {
            int wrow = waveN * 64 + sn * 16 + (lane & 15);
            int boff = (lane >> 4) * 8;                 // B layout: two 8-half runs
            wf[sn].q[0] = *(const uint4*)&Wlds[cur][wrow * 40 + boff];
            wf[sn].q[1] = *(const uint4*)&Wlds[cur][wrow * 40 + boff + 16];
        }
        #pragma unroll
        for (int sm = 0; sm < 2; ++sm)
            #pragma unroll
            for (int sn = 0; sn < 4; ++sn)
                acc[sm][sn] = __builtin_amdgcn_wmma_f32_16x16x32_bf16(
                    false, af[sm].v, false, wf[sn].v, (short)0, acc[sm][sn],
                    false, false);
    };

    const int nk = K >> 5;
    stage(0, 0);
    int kk = 0;
    for (; kk + 1 < nk; ++kk) {               // steady state: branch-free body
        stage((kk + 1) & 1, kk + 1);          // 4 newer loads outstanding
        {   // branchless clamped prefetch of the k+2 tile
            int kp = kk + 2 < nk ? kk + 2 : nk - 1;
            __builtin_prefetch(&A[(size_t)(bm + (tid >> 1)) * K + (kp << 5)], 0, 3);
            __builtin_prefetch(&W[(size_t)(bn + (tid >> 1)) * K + (kp << 5)], 0, 3);
        }
        WAIT_ASYNCCNT(0x4);                   // tile kk complete (in-order)
        __syncthreads();
        compute(kk & 1);
        __syncthreads();                      // readers done before buffer reuse
    }
    WAIT_ASYNCCNT(0x0);                       // epilogue iteration
    __syncthreads();
    compute(kk & 1);

    // epilogue (C layout: col = lane&15, row = e + 8*(lane>=16))
    #pragma unroll
    for (int sm = 0; sm < 2; ++sm) {
        #pragma unroll
        for (int sn = 0; sn < 4; ++sn) {
            int gcol = bn + waveN * 64 + sn * 16 + (lane & 15);
            float bv = (mode == 1) ? 0.0f : bias[gcol];
            #pragma unroll
            for (int e = 0; e < 8; ++e) {
                int grow = bm + waveM * 32 + sm * 16 + e + ((lane >> 4) << 3);
                float v = acc[sm][sn][e] + bv;
                size_t idx = (size_t)grow * N + gcol;
                if (mode == 0) {
                    outf[idx] = v;
                } else if (mode == 1) {
                    outbf[idx] = f32_to_bf16(v);
                } else if (mode == 3) {
                    float g = 0.5f * v * (1.0f + erff(v * 0.70710678118654752f));
                    outbf[idx] = f32_to_bf16(g);
                } else { // 2, 4: resid + gate * v
                    float gate = mod[(size_t)grow * 3072 + modoff + gcol];
                    outf[idx] = resid[idx] + gate * v;
                }
            }
        }
    }
}

// ---------------------------------------------------------------------------
// Flash attention: one block per (b, h, 64-row q-block), 4 waves, wave=16 rows.
// S = Q K^T via WMMA, online softmax in C-layout registers, P through per-wave
// LDS to get A layout, O += P V via WMMA against V^T staged in LDS.
// K tile staged by async DMA; V transposed through VGPRs into LDS.
// ---------------------------------------------------------------------------
__global__ __launch_bounds__(128) void attn_kernel(
    const unsigned short* __restrict__ qkv, unsigned short* __restrict__ outbf)
{
    __shared__ unsigned short Klds[64 * 72];       // [key][d], +8 pad
    __shared__ unsigned short Vt[64 * 72];         // [d][key], +8 pad
    __shared__ unsigned short Plds[4 * 16 * 72];   // per-wave 16x64 P tile

    const int tid  = threadIdx.x;
    const int lane = tid & 31;
    const int wave = tid >> 5;
    const int qt = blockIdx.x;          // 0..31
    const int h  = blockIdx.y;          // 0..7
    const int b  = blockIdx.z;          // 0..3
    const int qbase = qt * 64;
    const size_t rowbase = (size_t)b * 2048;

    // preload Q fragments (constant over k-loop)
    FragU qf[2];
    {
        int qrow = qbase + wave * 16 + (lane & 15);
        const unsigned short* qp = qkv + (rowbase + qrow) * 1536 + h * 64;
        int base = (lane >> 4) * 8;
        #pragma unroll
        for (int j = 0; j < 2; ++j) {
            qf[j].q[0] = *(const uint4*)(qp + j * 32 + base);
            qf[j].q[1] = *(const uint4*)(qp + j * 32 + base + 16);
        }
    }

    floatx8 O[4];
    #pragma unroll
    for (int nt = 0; nt < 4; ++nt) O[nt] = {};
    float rowmax[8], rowsum[8];
    #pragma unroll
    for (int e = 0; e < 8; ++e) { rowmax[e] = -3.0e38f; rowsum[e] = 0.0f; }

    unsigned short* myP = Plds + wave * 16 * 72;
    const int nkb = qt + 1;   // causal: only k-blocks at/below the diagonal
    for (int kb = 0; kb < nkb; ++kb) {
        const int kbase = kb * 64;
        __syncthreads();   // protect K/Vt from previous iteration's readers
        // stage K via async DMA, V transposed via VGPRs
        #pragma unroll
        for (int rp = 0; rp < 4; ++rp) {
            int u = rp * 128 + tid;
            int row = u >> 3, seg = u & 7;
            const unsigned short* kp =
                qkv + (rowbase + kbase + row) * 1536 + 512 + h * 64 + seg * 8;
            async_load_b128(kp, lds_off(&Klds[row * 72 + seg * 8]));
            const unsigned short* vp =
                qkv + (rowbase + kbase + row) * 1536 + 1024 + h * 64 + seg * 8;
            __align__(16) unsigned short tmp[8];
            *(uint4*)tmp = *(const uint4*)vp;
            #pragma unroll
            for (int jj = 0; jj < 8; ++jj)
                Vt[(seg * 8 + jj) * 72 + row] = tmp[jj];
        }
        WAIT_ASYNCCNT(0x0);
        __syncthreads();

        // S = Q K^T  (4 key-subtiles of 16, K-dim 64 = 2 WMMAs each)
        floatx8 sbuf[4];
        #pragma unroll
        for (int ks = 0; ks < 4; ++ks) {
            floatx8 s = {};
            #pragma unroll
            for (int j = 0; j < 2; ++j) {
                FragU kf;
                int n = ks * 16 + (lane & 15);
                int boff = n * 72 + j * 32 + (lane >> 4) * 8;
                kf.q[0] = *(const uint4*)&Klds[boff];
                kf.q[1] = *(const uint4*)&Klds[boff + 16];
                s = __builtin_amdgcn_wmma_f32_16x16x32_bf16(
                    false, qf[j].v, false, kf.v, (short)0, s, false, false);
            }
            int key = kbase + ks * 16 + (lane & 15);
            #pragma unroll
            for (int e = 0; e < 8; ++e) {
                int qrow = qbase + wave * 16 + e + ((lane >> 4) << 3);
                float sv = s[e] * 0.125f;               // DH^-0.5
                s[e] = (key <= qrow) ? sv : -3.0e38f;   // causal mask
            }
            sbuf[ks] = s;
        }

        // online softmax (row stats replicated across each 16-lane group)
        float mnew[8];
        #pragma unroll
        for (int e = 0; e < 8; ++e) mnew[e] = rowmax[e];
        #pragma unroll
        for (int ks = 0; ks < 4; ++ks)
            #pragma unroll
            for (int e = 0; e < 8; ++e) mnew[e] = fmaxf(mnew[e], sbuf[ks][e]);
        #pragma unroll
        for (int off = 8; off >= 1; off >>= 1)
            #pragma unroll
            for (int e = 0; e < 8; ++e)
                mnew[e] = fmaxf(mnew[e], __shfl_xor(mnew[e], off));

        float fac[8], psum[8];
        #pragma unroll
        for (int e = 0; e < 8; ++e) {
            fac[e] = __expf(rowmax[e] - mnew[e]);
            rowmax[e] = mnew[e];
            psum[e] = 0.0f;
        }
        #pragma unroll
        for (int nt = 0; nt < 4; ++nt)
            #pragma unroll
            for (int e = 0; e < 8; ++e) O[nt][e] *= fac[e];

        #pragma unroll
        for (int ks = 0; ks < 4; ++ks)
            #pragma unroll
            for (int e = 0; e < 8; ++e) {
                float pv = __expf(sbuf[ks][e] - mnew[e]);
                psum[e] += pv;
                int prow = e + ((lane >> 4) << 3);
                myP[prow * 72 + ks * 16 + (lane & 15)] = f32_to_bf16(pv);
            }
        #pragma unroll
        for (int off = 8; off >= 1; off >>= 1)
            #pragma unroll
            for (int e = 0; e < 8; ++e) psum[e] += __shfl_xor(psum[e], off);
        #pragma unroll
        for (int e = 0; e < 8; ++e) rowsum[e] = rowsum[e] * fac[e] + psum[e];

        __syncthreads();   // P visible across lanes

        // O += P V  (K-dim = 64 keys = 2 WMMAs, 4 dh-subtiles)
        #pragma unroll
        for (int j = 0; j < 2; ++j) {
            FragU pf;
            int m = lane & 15;
            int base = (lane >> 4) * 8;
            pf.q[0] = *(const uint4*)&myP[m * 72 + j * 32 + base];
            pf.q[1] = *(const uint4*)&myP[m * 72 + j * 32 + base + 16];
            #pragma unroll
            for (int nt = 0; nt < 4; ++nt) {
                FragU vf;
                int d = nt * 16 + (lane & 15);
                int off = d * 72 + j * 32 + (lane >> 4) * 8;
                vf.q[0] = *(const uint4*)&Vt[off];
                vf.q[1] = *(const uint4*)&Vt[off + 16];
                O[nt] = __builtin_amdgcn_wmma_f32_16x16x32_bf16(
                    false, pf.v, false, vf.v, (short)0, O[nt], false, false);
            }
        }
    }

    // normalize and write bf16 attention output [B*T, H*DH]
    #pragma unroll
    for (int e = 0; e < 8; ++e) {
        float inv = 1.0f / rowsum[e];
        int qrow = qbase + wave * 16 + e + ((lane >> 4) << 3);
        unsigned short* op = outbf + (rowbase + qrow) * 512 + h * 64;
        #pragma unroll
        for (int nt = 0; nt < 4; ++nt)
            op[nt * 16 + (lane & 15)] = f32_to_bf16(O[nt][e] * inv);
    }
}

// ---------------------------------------------------------------------------
// Host-side launch
// ---------------------------------------------------------------------------
extern "C" void kernel_launch(void* const* d_in, const int* in_sizes, int n_in,
                              void* d_out, int out_size, void* d_ws, size_t ws_size,
                              hipStream_t stream) {
    (void)in_sizes; (void)n_in; (void)out_size; (void)ws_size;
    const float* x    = (const float*)d_in[0];
    const float* ae   = (const float*)d_in[1];
    // d_in[2] = causal_mask (unused; causality handled analytically)
    const float* Wqkv = (const float*)d_in[3];
    const float* Wout = (const float*)d_in[4];
    const float* bout = (const float*)d_in[5];
    const float* W1   = (const float*)d_in[6];
    const float* b1   = (const float*)d_in[7];
    const float* W2   = (const float*)d_in[8];
    const float* b2   = (const float*)d_in[9];
    const float* Wmod = (const float*)d_in[10];
    const float* bmod = (const float*)d_in[11];
    float* out = (float*)d_out;

    const int M = 8192;   // B*T

    char* p = (char*)d_ws;
    auto alloc = [&](size_t bytes) {
        char* r = p;
        p += (bytes + 255) & ~(size_t)255;
        return (void*)r;
    };
    unsigned short* wmod_bf = (unsigned short*)alloc((size_t)3072 * 256 * 2);
    unsigned short* wqkv_bf = (unsigned short*)alloc((size_t)1536 * 512 * 2);
    unsigned short* wout_bf = (unsigned short*)alloc((size_t)512 * 512 * 2);
    unsigned short* w1_bf   = (unsigned short*)alloc((size_t)2048 * 512 * 2);
    unsigned short* w2_bf   = (unsigned short*)alloc((size_t)512 * 2048 * 2);
    unsigned short* asilu   = (unsigned short*)alloc((size_t)M * 256 * 2);
    float*          modf    = (float*)alloc((size_t)M * 3072 * 4);
    unsigned short* h1bf    = (unsigned short*)alloc((size_t)M * 512 * 2);
    unsigned short* qkvbf   = (unsigned short*)alloc((size_t)M * 1536 * 2);
    unsigned short* attnbf  = (unsigned short*)alloc((size_t)M * 512 * 2);
    float*          x2f     = (float*)alloc((size_t)M * 512 * 4);
    unsigned short* h2bf    = (unsigned short*)alloc((size_t)M * 512 * 2);
    unsigned short* ubf     = (unsigned short*)alloc((size_t)M * 2048 * 2);

    auto cvt = [&](const float* src, unsigned short* dst, int n) {
        f2bf_kernel<<<(n + 255) / 256, 256, 0, stream>>>(src, dst, n);
    };
    cvt(Wmod, wmod_bf, 3072 * 256);
    cvt(Wqkv, wqkv_bf, 1536 * 512);
    cvt(Wout, wout_bf, 512 * 512);
    cvt(W1,   w1_bf,   2048 * 512);
    cvt(W2,   w2_bf,   512 * 2048);

    silu_bf16_kernel<<<(M * 256 + 255) / 256, 256, 0, stream>>>(ae, asilu, M * 256);

    // mod = silu(ae) @ Wmod^T + bmod        [8192 x 3072]
    gemm_bf16_kernel<<<dim3(M / 128, 3072 / 128), 256, 0, stream>>>(
        asilu, wmod_bf, bmod, 256, 3072, 0, nullptr, nullptr, 0, modf, nullptr);

    // h1 = LN(x)*(1+scale1)+shift1
    adaln_kernel<<<M / 8, 256, 0, stream>>>(x, modf, 512, 0, h1bf);

    // qkv = h1 @ Wqkv^T                     [8192 x 1536]
    gemm_bf16_kernel<<<dim3(M / 128, 1536 / 128), 256, 0, stream>>>(
        h1bf, wqkv_bf, nullptr, 512, 1536, 1, nullptr, nullptr, 0, nullptr, qkvbf);

    // attention
    attn_kernel<<<dim3(32, 8, 4), 128, 0, stream>>>(qkvbf, attnbf);

    // x2 = x + gate1 * (attn @ Wout^T + bout)
    gemm_bf16_kernel<<<dim3(M / 128, 512 / 128), 256, 0, stream>>>(
        attnbf, wout_bf, bout, 512, 512, 2, x, modf, 1024, x2f, nullptr);

    // h2 = LN(x2)*(1+scale2)+shift2
    adaln_kernel<<<M / 8, 256, 0, stream>>>(x2f, modf, 2048, 1536, h2bf);

    // u = gelu(h2 @ W1^T + b1)              [8192 x 2048]
    gemm_bf16_kernel<<<dim3(M / 128, 2048 / 128), 256, 0, stream>>>(
        h2bf, w1_bf, b1, 512, 2048, 3, nullptr, nullptr, 0, nullptr, ubf);

    // out = x2 + gate2 * (u @ W2^T + b2)
    gemm_bf16_kernel<<<dim3(M / 128, 512 / 128), 256, 0, stream>>>(
        ubf, w2_bf, b2, 2048, 512, 4, x2f, modf, 2560, out, nullptr);
}
